// RelativePositionalAttention_24464133718913
// MI455X (gfx1250) — matrix operand order repaired and checked by reading, not compiled, not run on hardware
//
#include <hip/hip_runtime.h>
#include <hip/hip_bf16.h>

typedef __attribute__((ext_vector_type(16))) _Float16 v16h;
typedef __attribute__((ext_vector_type(8)))  _Float16 v8h;
typedef __attribute__((ext_vector_type(8)))  float    v8f;
typedef __attribute__((__vector_size__(4 * sizeof(int)))) int v4i_b;

#define LSEQ 1024
#define DMODEL 1024
#define NHEAD 16
#define DHEAD 64
#define NORMC 0.125f /* 1/sqrt(64) */

// ---------------------------------------------------------------------------
// CDNA5 async global->LDS copy (guarded; falls back to plain vector copy)
// ---------------------------------------------------------------------------
#if defined(__gfx1250__) &&                                                    \
    __has_builtin(__builtin_amdgcn_global_load_async_to_lds_b128) &&           \
    __has_builtin(__builtin_amdgcn_s_wait_asynccnt)
#define ASYNC_CP16(dst_lds, src_glob)                                          \
  __builtin_amdgcn_global_load_async_to_lds_b128(                              \
      (__attribute__((address_space(1))) v4i_b*)(src_glob),                    \
      (__attribute__((address_space(3))) v4i_b*)(dst_lds), 0, 0)
#define ASYNC_WAIT() __builtin_amdgcn_s_wait_asynccnt(0)
#else
#define ASYNC_CP16(dst_lds, src_glob)                                          \
  (*(uint4*)(dst_lds) = *(const uint4*)(src_glob))
#define ASYNC_WAIT()
#endif

// ---------------------------------------------------------------------------
// WMMA fragment helpers (ISA 7.12.2 layouts, wave32) -- all b128 LDS loads
// ---------------------------------------------------------------------------
__device__ __forceinline__ v16h load_a16v(const _Float16* base, int stride, int lane) {
  const _Float16* p = base + (lane & 15) * stride + (lane >> 4) * 8;
  v8h lo = *(const v8h*)(p);
  v8h hi = *(const v8h*)(p + 16);
  return __builtin_shufflevector(lo, hi, 0, 1, 2, 3, 4, 5, 6, 7, 8, 9, 10, 11,
                                 12, 13, 14, 15);
}

__device__ __forceinline__ v16h load_bT16v(const _Float16* base, int stride, int lane) {
  const _Float16* p = base + (lane & 15) * stride + (lane >> 4) * 8;
  v8h lo = *(const v8h*)(p);
  v8h hi = *(const v8h*)(p + 16);
  return __builtin_shufflevector(lo, hi, 0, 1, 2, 3, 4, 5, 6, 7, 8, 9, 10, 11,
                                 12, 13, 14, 15);
}

#define WMMA_F16(a, b, c) \
  __builtin_amdgcn_wmma_f32_16x16x32_f16(false, (a), false, (b), (short)0, (c), false, false)

// ---------------------------------------------------------------------------
// Kernel 0a: bulk f32 -> f16 convert (8 elems/thread, vectorized)
// ---------------------------------------------------------------------------
__global__ __launch_bounds__(256) void cvt_f32_f16(const float* __restrict__ X,
                                                   _Float16* __restrict__ Y) {
  const size_t i = ((size_t)blockIdx.x * 256 + threadIdx.x) * 8;
  const float4 a = *(const float4*)(X + i);
  const float4 b = *(const float4*)(X + i + 4);
  v8h o;
  o[0] = (_Float16)a.x; o[1] = (_Float16)a.y;
  o[2] = (_Float16)a.z; o[3] = (_Float16)a.w;
  o[4] = (_Float16)b.x; o[5] = (_Float16)b.y;
  o[6] = (_Float16)b.z; o[7] = (_Float16)b.w;
  *(v8h*)(Y + i) = o;
}

// ---------------------------------------------------------------------------
// Kernel 0b: weight pre-transpose + f32->f16   Wt[col][k] = W[k][col]
// rk_mode=1: W is r_kernel (NHEAD, K, 64): Wt[col][k] = W[col>>6][k][col&63]
// ---------------------------------------------------------------------------
__global__ __launch_bounds__(256) void transpose_w_f16(
    const float* __restrict__ W, _Float16* __restrict__ Wt, int K, int N,
    int rk_mode) {
  const int c0 = blockIdx.x * 64, k0 = blockIdx.y * 64;
  for (int idx = threadIdx.x; idx < 64 * 64; idx += 256) {
    int c = idx & 63, kk = idx >> 6;  // coalesced read along c
    int col = c0 + c, k = k0 + kk;
    float val = rk_mode ? W[((size_t)(col >> 6) * K + k) * 64 + (col & 63)]
                        : W[(size_t)k * N + col];
    Wt[(size_t)col * K + k] = (_Float16)val;
  }
}

// ---------------------------------------------------------------------------
// Kernel 1: Y(f16)[MxN] = Xh(f16)[MxK] @ Wt(f16,N-major)[NxK] (+ colbias)
// Double-buffered: async tile k+1 in flight while WMMAs consume tile k.
// vt_mode=1: write Y transposed per head: Y[((b*16+n)*64+h)*LSEQ + l]
// ---------------------------------------------------------------------------
__global__ __launch_bounds__(128) void gemm_ff(
    const _Float16* __restrict__ Xh, const _Float16* __restrict__ Wt,
    const float* __restrict__ bias, _Float16* __restrict__ Y, int M, int K,
    int N, int vt_mode) {
  alignas(16) __shared__ _Float16 Xs[2][64 * 64];
  alignas(16) __shared__ _Float16 Ws[2][64 * 64];
  const int tid = threadIdx.x, lane = tid & 31, wv = tid >> 5;
  const int n0 = blockIdx.x * 64;
  const int m0 = blockIdx.y * 64;

  v8f acc[4];
  const v8f z8 = {0.f, 0.f, 0.f, 0.f, 0.f, 0.f, 0.f, 0.f};
#pragma unroll
  for (int c = 0; c < 4; ++c) acc[c] = z8;

  auto stage = [&](int buf, int k0) {
#pragma unroll
    for (int it = 0; it < 4; ++it) {
      int idx = tid + 128 * it;
      int row = idx >> 3, cc = (idx & 7) * 8;
      ASYNC_CP16(&Xs[buf][row * 64 + cc],
                 Xh + (size_t)(m0 + row) * K + k0 + cc);
      ASYNC_CP16(&Ws[buf][row * 64 + cc],
                 Wt + (size_t)(n0 + row) * K + k0 + cc);
    }
  };

  const int KT = K / 64;
  stage(0, 0);
  ASYNC_WAIT();
  __syncthreads();
  for (int kt = 0; kt < KT; ++kt) {
    const int cur = kt & 1;
    if (kt + 1 < KT) stage(cur ^ 1, (kt + 1) * 64);  // overlap with compute
#pragma unroll
    for (int ks = 0; ks < 2; ++ks) {
      v16h a = load_a16v(&Xs[cur][(wv * 16) * 64 + ks * 32], 64, lane);
#pragma unroll
      for (int c = 0; c < 4; ++c) {
        v16h b = load_bT16v(&Ws[cur][(c * 16) * 64 + ks * 32], 64, lane);
        acc[c] = WMMA_F16(a, b, acc[c]);
      }
    }
    ASYNC_WAIT();
    __syncthreads();
  }
  // ---- epilogue: C-frag -> LDS -> coalesced b128 global stores ----
  const int h2 = lane >> 4, nn = lane & 15;
  _Float16* Ot = &Xs[0][0];
#pragma unroll
  for (int c = 0; c < 4; ++c)
#pragma unroll
    for (int j = 0; j < 8; ++j) {
      float v = acc[c][j];
      if (bias) v += bias[n0 + c * 16 + nn];
      if (!vt_mode)
        Ot[(wv * 16 + j + 8 * h2) * 64 + c * 16 + nn] = (_Float16)v;
      else  // transposed: row = h (col tile), col = l (row tile)
        Ot[(c * 16 + nn) * 64 + wv * 16 + j + 8 * h2] = (_Float16)v;
    }
  __syncthreads();
  if (!vt_mode) {
#pragma unroll
    for (int it = 0; it < 4; ++it) {
      int idx = tid + 128 * it;
      int row = idx >> 3, cc = (idx & 7) * 8;
      *(uint4*)(Y + (size_t)(m0 + row) * N + n0 + cc) =
          *(uint4*)(Ot + row * 64 + cc);
    }
  } else {
    const int bb = m0 >> 10, head = n0 >> 6, lbase = m0 & 1023;
#pragma unroll
    for (int it = 0; it < 4; ++it) {
      int idx = tid + 128 * it;
      int row = idx >> 3, cc = (idx & 7) * 8;  // row = h
      *(uint4*)(Y + (((size_t)bb * NHEAD + head) * DHEAD + row) * LSEQ + lbase + cc) =
          *(uint4*)(Ot + row * 64 + cc);
    }
  }
}

// ---------------------------------------------------------------------------
// Kernel 2: fused relative attention for one (b, head, 64-query tile)
// score[l,j] = (qw[l].kh[j] + qr[l].posk[L + j - l]) * NORM - 1e6*mask[j]
// ---------------------------------------------------------------------------
__global__ __launch_bounds__(128) void attn_kernel(
    const _Float16* __restrict__ qh, const _Float16* __restrict__ kh,
    const _Float16* __restrict__ vT, const _Float16* __restrict__ posk,
    const float* __restrict__ rwb, const float* __restrict__ rrb,
    const int* __restrict__ mask, _Float16* __restrict__ O) {
  alignas(16) __shared__ _Float16 Kt[64 * 64];      // [j][h]
  alignas(16) __shared__ _Float16 Vt[64 * 64];      // [h][j]  (transposed V)
  alignas(16) __shared__ _Float16 Band[128 * 64];   // [i][h]; Q staging in prologue
  alignas(16) __shared__ _Float16 pscr[4][16 * 128];
  __shared__ float maskf[LSEQ];

  const int tid = threadIdx.x, lane = tid & 31, wv = tid >> 5;
  const int l0 = blockIdx.x * 64;
  const int n = blockIdx.y;
  const int b = blockIdx.z;
  const size_t rowbase = (size_t)(b * LSEQ) * DMODEL + n * DHEAD;
  const size_t vtbase = ((size_t)b * NHEAD + n) * DHEAD * LSEQ;
  const int h2 = lane >> 4, nn = lane & 15;

  for (int i = tid; i < LSEQ; i += 128)
    maskf[i] = 1.0e6f * (float)mask[b * LSEQ + i];

  // ---- prologue: Q+r_w_bias / Q+r_r_bias A-fragments held in VGPRs ----
  for (int idx = tid; idx < 64 * 64; idx += 128) {
    int r = idx >> 6, c = idx & 63;
    float qv = (float)qh[rowbase + (size_t)(l0 + r) * DMODEL + c];
    Band[idx] = (_Float16)(qv + rwb[n * DHEAD + c]);
  }
  __syncthreads();
  v16h aw0 = load_a16v(Band + (wv * 16) * 64, 64, lane);
  v16h aw1 = load_a16v(Band + (wv * 16) * 64 + 32, 64, lane);
  __syncthreads();
  for (int idx = tid; idx < 64 * 64; idx += 128) {
    int r = idx >> 6, c = idx & 63;
    float qv = (float)qh[rowbase + (size_t)(l0 + r) * DMODEL + c];
    Band[idx] = (_Float16)(qv + rrb[n * DHEAD + c]);
  }
  __syncthreads();
  v16h ar0 = load_a16v(Band + (wv * 16) * 64, 64, lane);
  v16h ar1 = load_a16v(Band + (wv * 16) * 64 + 32, 64, lane);

  const v8f z8 = {0.f, 0.f, 0.f, 0.f, 0.f, 0.f, 0.f, 0.f};
  v8f Oacc[4];
#pragma unroll
  for (int c = 0; c < 4; ++c) Oacc[c] = z8;
  float m_run[8], s_run[8];
#pragma unroll
  for (int j = 0; j < 8; ++j) { m_run[j] = -1.0e30f; s_run[j] = 0.f; }

  for (int jt = 0; jt < LSEQ / 64; ++jt) {
    const int j0 = jt * 64;
    __syncthreads();
    // ---- async-stage K[j][h], V^T[h][j], pos band[i][h] ----
    for (int ch = tid; ch < 512; ch += 128) {
      int r = ch >> 3, cc = (ch & 7) * 8;
      ASYNC_CP16(Kt + r * 64 + cc,
                 kh + rowbase + (size_t)(j0 + r) * DMODEL + cc);
      ASYNC_CP16(Vt + r * 64 + cc,
                 vT + vtbase + (size_t)r * LSEQ + j0 + cc);
    }
    const int rb = LSEQ + j0 - l0 - 63;  // band row base: r = L + j - l
    for (int ch = tid; ch < 1024; ch += 128) {
      int i = ch >> 3, cc = (ch & 7) * 8;
      int r = rb + i;
      if (r < 2 * LSEQ)
        ASYNC_CP16(Band + i * 64 + cc,
                   posk + (size_t)r * DMODEL + n * DHEAD + cc);
      else
        *(uint4*)(Band + i * 64 + cc) = make_uint4(0, 0, 0, 0);
    }
    ASYNC_WAIT();
    __syncthreads();

    // ---- scores S = Qw @ K^T ; band P = Qr @ Band^T ----
    v8f S4[4], P4[8];
#pragma unroll
    for (int c = 0; c < 4; ++c) S4[c] = z8;
#pragma unroll
    for (int c = 0; c < 8; ++c) P4[c] = z8;
#pragma unroll
    for (int ks = 0; ks < 2; ++ks) {
      v16h a = ks ? aw1 : aw0;
#pragma unroll
      for (int c = 0; c < 4; ++c) {
        v16h bf = load_bT16v(Kt + (c * 16) * 64 + ks * 32, 64, lane);
        S4[c] = WMMA_F16(a, bf, S4[c]);
      }
      v16h ap = ks ? ar1 : ar0;
#pragma unroll
      for (int c = 0; c < 8; ++c) {
        v16h bf = load_bT16v(Band + (c * 16) * 64 + ks * 32, 64, lane);
        P4[c] = WMMA_F16(ap, bf, P4[c]);
      }
    }
    // ---- spill band product, gather rel-shift diagonal band ----
    _Float16* pw = pscr[wv];
#pragma unroll
    for (int c = 0; c < 8; ++c)
#pragma unroll
      for (int j = 0; j < 8; ++j)
        pw[(j + 8 * h2) * 128 + c * 16 + nn] = (_Float16)P4[c][j];
#pragma unroll
    for (int c = 0; c < 4; ++c)
#pragma unroll
      for (int j = 0; j < 8; ++j) {
        int r = j + 8 * h2;
        int jj = c * 16 + nn;
        int bi = jj + 63 - (wv * 16 + r);
        S4[c][j] = (S4[c][j] + (float)pw[r * 128 + bi]) * NORMC - maskf[j0 + jj];
      }
    // ---- online softmax (row stats live in 16-lane halves) ----
#pragma unroll
    for (int j = 0; j < 8; ++j) {
      float tm = fmaxf(fmaxf(S4[0][j], S4[1][j]), fmaxf(S4[2][j], S4[3][j]));
      tm = fmaxf(tm, __shfl_xor(tm, 1, 32));
      tm = fmaxf(tm, __shfl_xor(tm, 2, 32));
      tm = fmaxf(tm, __shfl_xor(tm, 4, 32));
      tm = fmaxf(tm, __shfl_xor(tm, 8, 32));
      float mn = fmaxf(m_run[j], tm);
      float fs = __expf(m_run[j] - mn);
      float ps = 0.f;
#pragma unroll
      for (int c = 0; c < 4; ++c) {
        float e = __expf(S4[c][j] - mn);
        S4[c][j] = e;
        ps += e;
      }
      ps += __shfl_xor(ps, 1, 32);
      ps += __shfl_xor(ps, 2, 32);
      ps += __shfl_xor(ps, 4, 32);
      ps += __shfl_xor(ps, 8, 32);
      s_run[j] = s_run[j] * fs + ps;
      m_run[j] = mn;
#pragma unroll
      for (int c = 0; c < 4; ++c) Oacc[c][j] = Oacc[c][j] * fs;
    }
    // ---- probs -> f16 A-frags via LDS, O += P @ V ----
#pragma unroll
    for (int c = 0; c < 4; ++c)
#pragma unroll
      for (int j = 0; j < 8; ++j)
        pw[(j + 8 * h2) * 64 + c * 16 + nn] = (_Float16)S4[c][j];
#pragma unroll
    for (int ks = 0; ks < 2; ++ks) {
      v16h ap = load_a16v(pw + ks * 32, 64, lane);
#pragma unroll
      for (int c = 0; c < 4; ++c) {
        v16h bv2 = load_bT16v(Vt + (c * 16) * 64 + ks * 32, 64, lane);
        Oacc[c] = WMMA_F16(ap, bv2, Oacc[c]);
      }
    }
  }
  // ---- finalize: normalize, stage via LDS, coalesced stores ----
  float inv[8];
#pragma unroll
  for (int j = 0; j < 8; ++j) inv[j] = 1.0f / s_run[j];
  __syncthreads();
#pragma unroll
  for (int c = 0; c < 4; ++c)
#pragma unroll
    for (int j = 0; j < 8; ++j)
      Kt[(wv * 16 + j + 8 * h2) * 64 + c * 16 + nn] =
          (_Float16)(Oacc[c][j] * inv[j]);
  __syncthreads();
#pragma unroll
  for (int it = 0; it < 4; ++it) {
    int idx = tid + 128 * it;
    int row = idx >> 3, cc = (idx & 7) * 8;
    *(uint4*)(O + (size_t)(b * LSEQ + l0 + row) * DMODEL + n * DHEAD + cc) =
        *(uint4*)(Kt + row * 64 + cc);
  }
}

// ---------------------------------------------------------------------------
// Kernel 3a: res(f32) = Xh(f16) @ Wot(f16,N-major) + bo + residual(q)
// Double-buffered async staging, same pipeline as gemm_ff.
// ---------------------------------------------------------------------------
__global__ __launch_bounds__(128) void gemm_out(
    const _Float16* __restrict__ Xh, const _Float16* __restrict__ Wt,
    const float* __restrict__ bo, const float* __restrict__ resid,
    float* __restrict__ Y, int M, int K, int N) {
  alignas(16) __shared__ _Float16 Xs[2][64 * 64];  // also f32 out staging (16KB)
  alignas(16) __shared__ _Float16 Ws[2][64 * 64];
  const int tid = threadIdx.x, lane = tid & 31, wv = tid >> 5;
  const int n0 = blockIdx.x * 64;
  const int m0 = blockIdx.y * 64;

  v8f acc[4];
  const v8f z8 = {0.f, 0.f, 0.f, 0.f, 0.f, 0.f, 0.f, 0.f};
#pragma unroll
  for (int c = 0; c < 4; ++c) acc[c] = z8;

  auto stage = [&](int buf, int k0) {
#pragma unroll
    for (int it = 0; it < 4; ++it) {
      int idx = tid + 128 * it;
      int row = idx >> 3, cc = (idx & 7) * 8;
      ASYNC_CP16(&Xs[buf][row * 64 + cc],
                 Xh + (size_t)(m0 + row) * K + k0 + cc);
      ASYNC_CP16(&Ws[buf][row * 64 + cc],
                 Wt + (size_t)(n0 + row) * K + k0 + cc);
    }
  };

  const int KT = K / 64;
  stage(0, 0);
  ASYNC_WAIT();
  __syncthreads();
  for (int kt = 0; kt < KT; ++kt) {
    const int cur = kt & 1;
    if (kt + 1 < KT) stage(cur ^ 1, (kt + 1) * 64);
#pragma unroll
    for (int ks = 0; ks < 2; ++ks) {
      v16h a = load_a16v(&Xs[cur][(wv * 16) * 64 + ks * 32], 64, lane);
#pragma unroll
      for (int c = 0; c < 4; ++c) {
        v16h b = load_bT16v(&Ws[cur][(c * 16) * 64 + ks * 32], 64, lane);
        acc[c] = WMMA_F16(a, b, acc[c]);
      }
    }
    ASYNC_WAIT();
    __syncthreads();
  }
  // ---- epilogue: +bo, stage f32 tile in LDS, fused residual + float4 stores
  const int h2 = lane >> 4, nn = lane & 15;
  float* Osta = (float*)&Xs[0][0];
#pragma unroll
  for (int c = 0; c < 4; ++c)
#pragma unroll
    for (int j = 0; j < 8; ++j)
      Osta[(wv * 16 + j + 8 * h2) * 64 + c * 16 + nn] =
          acc[c][j] + bo[n0 + c * 16 + nn];
  __syncthreads();
#pragma unroll
  for (int it = 0; it < 8; ++it) {
    int idx = tid + 128 * it;
    int row = idx >> 4, cc = (idx & 15) * 4;
    float4 o = *(float4*)(Osta + row * 64 + cc);
    const float4 r4 = *(const float4*)(resid + (size_t)(m0 + row) * N + n0 + cc);
    o.x += r4.x; o.y += r4.y; o.z += r4.z; o.w += r4.w;
    *(float4*)(Y + (size_t)(m0 + row) * N + n0 + cc) = o;
  }
}

// ---------------------------------------------------------------------------
// Kernel 3b: row-wise LayerNorm (1024 elems/row)
// ---------------------------------------------------------------------------
__global__ __launch_bounds__(256) void layernorm_k(
    const float* __restrict__ X, const float* __restrict__ w,
    const float* __restrict__ bp, float* __restrict__ out, int N) {
  __shared__ float red1[8];
  __shared__ float red2[8];
  const int row = blockIdx.x;
  const float* x = X + (size_t)row * N;
  const int tid = threadIdx.x;

  float s = 0.f;
  for (int i = tid; i < N; i += 256) s += x[i];
#pragma unroll
  for (int m = 1; m < 32; m <<= 1) s += __shfl_xor(s, m, 32);
  if ((tid & 31) == 0) red1[tid >> 5] = s;
  __syncthreads();
  float tot = 0.f;
#pragma unroll
  for (int k2 = 0; k2 < 8; ++k2) tot += red1[k2];
  const float mu = tot / (float)N;

  float d = 0.f;
  for (int i = tid; i < N; i += 256) { float t = x[i] - mu; d += t * t; }
#pragma unroll
  for (int m = 1; m < 32; m <<= 1) d += __shfl_xor(d, m, 32);
  if ((tid & 31) == 0) red2[tid >> 5] = d;
  __syncthreads();
  float vtot = 0.f;
#pragma unroll
  for (int k2 = 0; k2 < 8; ++k2) vtot += red2[k2];
  const float rstd = rsqrtf(vtot / (float)N + 1e-9f);

  for (int i = tid; i < N; i += 256)
    out[(size_t)row * N + i] = (x[i] - mu) * rstd * w[i] + bp[i];
}

// ---------------------------------------------------------------------------
extern "C" void kernel_launch(void* const* d_in, const int* in_sizes, int n_in,
                              void* d_out, int out_size, void* d_ws, size_t ws_size,
                              hipStream_t stream) {
  (void)in_sizes; (void)n_in; (void)out_size; (void)ws_size;
  const float* q    = (const float*)d_in[0];
  const float* k    = (const float*)d_in[1];
  const float* v    = (const float*)d_in[2];
  const float* pos  = (const float*)d_in[3];
  const float* Wq   = (const float*)d_in[4];
  const float* Wk   = (const float*)d_in[5];
  const float* bk   = (const float*)d_in[6];
  const float* Wv   = (const float*)d_in[7];
  const float* bv   = (const float*)d_in[8];
  const float* rwb  = (const float*)d_in[9];
  const float* rrb  = (const float*)d_in[10];
  const float* rker = (const float*)d_in[11];
  const float* Wo   = (const float*)d_in[12];
  const float* bo   = (const float*)d_in[13];
  const float* lnw  = (const float*)d_in[14];
  const float* lnb  = (const float*)d_in[15];
  const int*   msk  = (const int*)d_in[16];

  char* ws = (char*)d_ws;  // requires ~90 MB workspace
  _Float16* qhw  = (_Float16*)(ws);                        // 8 MB (4096x1024)
  _Float16* khw  = (_Float16*)(ws + ((size_t)8  << 20));   // 8 MB
  _Float16* vtw  = (_Float16*)(ws + ((size_t)16 << 20));   // 8 MB (b,n,h,l)
  _Float16* pkw  = (_Float16*)(ws + ((size_t)24 << 20));   // 4 MB (2048x1024)
  _Float16* ow   = (_Float16*)(ws + ((size_t)28 << 20));   // 8 MB
  float*    res  = (float*)   (ws + ((size_t)36 << 20));   // 16 MB
  _Float16* Wqt  = (_Float16*)(ws + ((size_t)52 << 20));   // 2 MB each
  _Float16* Wkt  = (_Float16*)(ws + ((size_t)54 << 20));
  _Float16* Wvt  = (_Float16*)(ws + ((size_t)56 << 20));
  _Float16* Wrt  = (_Float16*)(ws + ((size_t)58 << 20));
  _Float16* Wot  = (_Float16*)(ws + ((size_t)60 << 20));
  _Float16* qf   = (_Float16*)(ws + ((size_t)62 << 20));   // 8 MB f16 activations
  _Float16* kf   = (_Float16*)(ws + ((size_t)70 << 20));   // 8 MB
  _Float16* vf   = (_Float16*)(ws + ((size_t)78 << 20));   // 8 MB
  _Float16* posf = (_Float16*)(ws + ((size_t)86 << 20));   // 4 MB

  const dim3 blk(128);
  const int M = 4 * LSEQ;  // 4096

  // activation f32->f16 pre-pass (removes conversions from GEMM hot loops)
  cvt_f32_f16<<<dim3(2048), dim3(256), 0, stream>>>(q,   qf);
  cvt_f32_f16<<<dim3(2048), dim3(256), 0, stream>>>(k,   kf);
  cvt_f32_f16<<<dim3(2048), dim3(256), 0, stream>>>(v,   vf);
  cvt_f32_f16<<<dim3(1024), dim3(256), 0, stream>>>(pos, posf);

  // weight prep (one-time transpose + f16 convert)
  transpose_w_f16<<<dim3(16, 16), dim3(256), 0, stream>>>(Wq,   Wqt, DMODEL, DMODEL, 0);
  transpose_w_f16<<<dim3(16, 16), dim3(256), 0, stream>>>(Wk,   Wkt, DMODEL, DMODEL, 0);
  transpose_w_f16<<<dim3(16, 16), dim3(256), 0, stream>>>(Wv,   Wvt, DMODEL, DMODEL, 0);
  transpose_w_f16<<<dim3(16, 16), dim3(256), 0, stream>>>(rker, Wrt, DMODEL, DMODEL, 1);
  transpose_w_f16<<<dim3(16, 16), dim3(256), 0, stream>>>(Wo,   Wot, DMODEL, DMODEL, 0);

  gemm_ff<<<dim3(16, 64), blk, 0, stream>>>(qf,   Wqt, nullptr, qhw, M,    DMODEL, DMODEL, 0);
  gemm_ff<<<dim3(16, 64), blk, 0, stream>>>(kf,   Wkt, bk,      khw, M,    DMODEL, DMODEL, 0);
  gemm_ff<<<dim3(16, 64), blk, 0, stream>>>(vf,   Wvt, bv,      vtw, M,    DMODEL, DMODEL, 1);
  gemm_ff<<<dim3(16, 32), blk, 0, stream>>>(posf, Wrt, nullptr, pkw, 2048, DMODEL, DMODEL, 0);

  attn_kernel<<<dim3(LSEQ / 64, NHEAD, 4), blk, 0, stream>>>(qhw, khw, vtw, pkw,
                                                             rwb, rrb, msk, ow);

  gemm_out<<<dim3(16, 64), blk, 0, stream>>>(ow, Wot, bo, q, res, M, DMODEL, DMODEL);
  layernorm_k<<<dim3(M), dim3(256), 0, stream>>>(res, lnw, lnb, (float*)d_out, DMODEL);
}
